// CausalSelfAttention_65060164599984
// MI455X (gfx1250) — matrix operand order repaired and checked by reading, not compiled, not run on hardware
//
#include <hip/hip_runtime.h>

// ---------------------------------------------------------------------------
// CDNA5 (gfx1250) causal self-attention forward.
// All matmuls go through v_wmma_f32_16x16x32_bf16 (fp32 ref -> bf16 inputs,
// fp32 accumulate). Roofline: ~410 GFLOP vs ~0.3 GB HBM -> WMMA-bound, so
// everything is tiled for the matrix pipes; softmax is fused (flash-style)
// so the 4*16*2048^2 score matrix never leaves the WGP.
// ---------------------------------------------------------------------------

typedef __attribute__((ext_vector_type(16))) __bf16 v16bf;
typedef __attribute__((ext_vector_type(8)))  __bf16 v8bf;
typedef __attribute__((ext_vector_type(8)))  float  v8f;

#define B_  4
#define T_  2048
#define E_  2048
#define H_  16
#define HD_ 128

// ---------------------------------------------------------------------------
// fp32 -> bf16 conversion (RNE via clang __bf16 cast)
// ---------------------------------------------------------------------------
__global__ void f32_to_bf16_kernel(const float* __restrict__ in,
                                   __bf16* __restrict__ out, int n) {
  int i = blockIdx.x * blockDim.x + threadIdx.x;
  int stride = gridDim.x * blockDim.x;
  for (; i < n; i += stride) out[i] = (__bf16)in[i];
}

// ---------------------------------------------------------------------------
// C(MxN) = A(MxK) * B(NxK)^T, bf16 inputs, fp32 WMMA accumulate.
// Block tile 128x128, 8 waves (2x4), wave tile 64x32 (4x2 WMMA tiles).
// K-step 32. LDS tiles padded to stride 40 bf16 (80B rows: 20-dword stride,
// gcd(20,64)=4 -> 16 distinct banks across the 16 row-lanes, 16B aligned).
// Requires M%128==0, N%128==0, K%32==0 (true for all calls here).
// ---------------------------------------------------------------------------
template <typename OutT>
__global__ __launch_bounds__(256)
void gemm_bf16_tn(const __bf16* __restrict__ A, const __bf16* __restrict__ Bm,
                  OutT* __restrict__ C, int M, int N, int K) {
  __shared__ alignas(16) __bf16 As[128][40];
  __shared__ alignas(16) __bf16 Bs[128][40];

  const int tid  = threadIdx.x;
  const int wave = tid >> 5;
  const int lane = tid & 31;
  const int ln = lane & 15, lh = lane >> 4;
  const int m0 = blockIdx.y * 128;
  const int n0 = blockIdx.x * 128;
  const int wm = (wave >> 2) * 64;  // 0 / 64
  const int wn = (wave & 3) * 32;   // 0..96

  v8f acc[4][2];
#pragma unroll
  for (int mi = 0; mi < 4; ++mi)
#pragma unroll
    for (int ni = 0; ni < 2; ++ni) acc[mi][ni] = {};

  for (int k0 = 0; k0 < K; k0 += 32) {
    __syncthreads();
    // cooperative global -> LDS: 512 16B chunks per tile, 2 per thread
#pragma unroll
    for (int i = 0; i < 2; ++i) {
      int idx = tid + i * 256;
      int row = idx >> 2;
      int ko  = (idx & 3) * 8;
      *(v8bf*)&As[row][ko] = *(const v8bf*)&A[(size_t)(m0 + row) * K + k0 + ko];
      *(v8bf*)&Bs[row][ko] = *(const v8bf*)&Bm[(size_t)(n0 + row) * K + k0 + ko];
    }
    __syncthreads();

    // A-frag: lane<16 holds K{0..7,16..23}, lane>=16 K{8..15,24..31}
    v16bf af[4];
#pragma unroll
    for (int mi = 0; mi < 4; ++mi) {
      int r = wm + mi * 16 + ln;
      v8bf lo = *(const v8bf*)&As[r][lh * 8];
      v8bf hi = *(const v8bf*)&As[r][16 + lh * 8];
#pragma unroll
      for (int e = 0; e < 8; ++e) { af[mi][e] = lo[e]; af[mi][e + 8] = hi[e]; }
    }
    // B-frag: lane<16 holds K{0..7,16..23}, lane>=16 K{8..15,24..31}
    v16bf bfr[2];
#pragma unroll
    for (int ni = 0; ni < 2; ++ni) {
      int r = wn + ni * 16 + ln;
      v8bf lo = *(const v8bf*)&Bs[r][lh * 8];
      v8bf hi = *(const v8bf*)&Bs[r][16 + lh * 8];
#pragma unroll
      for (int e = 0; e < 8; ++e) { bfr[ni][e] = lo[e]; bfr[ni][e + 8] = hi[e]; }
    }
#pragma unroll
    for (int mi = 0; mi < 4; ++mi)
#pragma unroll
      for (int ni = 0; ni < 2; ++ni)
        acc[mi][ni] = __builtin_amdgcn_wmma_f32_16x16x32_bf16(
            false, af[mi], false, bfr[ni], (short)0, acc[mi][ni], false, false);
  }

  // C/D layout: N = lane&15, M = r + 8*(lane>>4)
#pragma unroll
  for (int mi = 0; mi < 4; ++mi)
#pragma unroll
    for (int ni = 0; ni < 2; ++ni) {
      int col = n0 + wn + ni * 16 + ln;
#pragma unroll
      for (int r = 0; r < 8; ++r) {
        int row = m0 + wm + mi * 16 + 8 * lh + r;
        C[(size_t)row * N + col] = (OutT)acc[mi][ni][r];
      }
    }
}

// ---------------------------------------------------------------------------
// RoPE + qkv split: [B*T][3E] bf16 -> q,k,v [B][H][T][HD] bf16 (rope on q,k)
// One thread per (b,h,t,j) rotation pair; pure-bandwidth (~200 MB).
// ---------------------------------------------------------------------------
__global__ void rope_split_kernel(const __bf16* __restrict__ qkv,
                                  const float* __restrict__ freqs,
                                  __bf16* __restrict__ q, __bf16* __restrict__ k,
                                  __bf16* __restrict__ v) {
  int idx = blockIdx.x * 256 + threadIdx.x;      // < B*H*T*64 = 2^23
  int j = idx & 63;
  int t = (idx >> 6) & (T_ - 1);
  int h = (idx >> 17) & (H_ - 1);
  int b = idx >> 21;

  float f  = freqs[t * 64 + j];
  float cs = __cosf(f), sn = __sinf(f);

  size_t ib = (size_t)(b * T_ + t) * (3 * E_) + h * HD_ + 2 * j;
  size_t ob = ((size_t)(b * H_ + h) * T_ + t) * HD_ + 2 * j;

  float q1 = (float)qkv[ib],          q2 = (float)qkv[ib + 1];
  float k1 = (float)qkv[ib + E_],     k2 = (float)qkv[ib + E_ + 1];
  float v1 = (float)qkv[ib + 2 * E_], v2 = (float)qkv[ib + 2 * E_ + 1];

  q[ob] = (__bf16)(q1 * cs - q2 * sn); q[ob + 1] = (__bf16)(q1 * sn + q2 * cs);
  k[ob] = (__bf16)(k1 * cs - k2 * sn); k[ob + 1] = (__bf16)(k1 * sn + k2 * cs);
  v[ob] = (__bf16)v1;                  v[ob + 1] = (__bf16)v2;
}

// ---------------------------------------------------------------------------
// Flash attention. Block = 128 queries of one (b,h); 8 waves x 16 queries.
// Waves share LDS K/V tiles (32 keys/iter). Per iter/wave: 8 WMMA for S,
// online softmax (shfl_xor over the 16-lane N dimension), P transposed
// C-layout -> A-layout through per-wave LDS, 8 WMMA for P@V.
// ---------------------------------------------------------------------------
__global__ __launch_bounds__(256)
void attn_kernel(const __bf16* __restrict__ Q, const __bf16* __restrict__ Kg,
                 const __bf16* __restrict__ Vg, __bf16* __restrict__ O) {
  __shared__ alignas(16) __bf16 Kt[32][136];    // [key][d], stride 136
  __shared__ alignas(16) __bf16 Vt[HD_][40];    // [d][key], transposed
  __shared__ alignas(16) __bf16 Ps[8][16][40];  // per-wave P staging

  const int tid  = threadIdx.x;
  const int wave = tid >> 5;
  const int lane = tid & 31;
  const int ln = lane & 15, lh = lane >> 4;
  const int bh   = blockIdx.y;                  // b*H + h
  const int qblk = blockIdx.x * 128;
  const int q0   = qblk + wave * 16;
  const size_t base = (size_t)bh * T_ * HD_;

  // Q fragments (A-layout), held in registers for the whole key loop
  v16bf qf[4];
#pragma unroll
  for (int c = 0; c < 4; ++c) {
    const __bf16* qp = Q + base + (size_t)(q0 + ln) * HD_ + c * 32 + lh * 8;
    v8bf lo = *(const v8bf*)qp;
    v8bf hi = *(const v8bf*)(qp + 16);
#pragma unroll
    for (int e = 0; e < 8; ++e) { qf[c][e] = lo[e]; qf[c][e + 8] = hi[e]; }
  }

  v8f oacc[8];
  float Mr[8], Lr[8];
#pragma unroll
  for (int d = 0; d < 8; ++d) oacc[d] = {};
#pragma unroll
  for (int r = 0; r < 8; ++r) { Mr[r] = -1e30f; Lr[r] = 0.f; }

  const float scale = 0.08838834764831845f;     // 1/sqrt(128)
  const int kend = qblk + 128;                  // uniform across the block

  for (int kb = 0; kb < kend; kb += 32) {
    __syncthreads();
    // cooperative K/V tile load; V transposed into Vt[d][key]
#pragma unroll
    for (int i = 0; i < 2; ++i) {
      int idx  = tid + i * 256;
      int key  = idx >> 4;
      int doff = (idx & 15) * 8;
      *(v8bf*)&Kt[key][doff] =
          *(const v8bf*)&Kg[base + (size_t)(kb + key) * HD_ + doff];
      v8bf vv = *(const v8bf*)&Vg[base + (size_t)(kb + key) * HD_ + doff];
#pragma unroll
      for (int e = 0; e < 8; ++e) Vt[doff + e][key] = vv[e];
    }
    __syncthreads();

    if (kb < q0 + 16) {                         // this wave's causal range
      // ---- S = Q * K^T, two 16-key tiles ----
      v8f s[2];
#pragma unroll
      for (int ti = 0; ti < 2; ++ti) {
        v8f sa = {};
#pragma unroll
        for (int c = 0; c < 4; ++c) {
          v8bf lo = *(const v8bf*)&Kt[ti * 16 + ln][c * 32 + lh * 8];
          v8bf hi = *(const v8bf*)&Kt[ti * 16 + ln][c * 32 + 16 + lh * 8];
          v16bf kf;
#pragma unroll
          for (int e = 0; e < 8; ++e) { kf[e] = lo[e]; kf[e + 8] = hi[e]; }
          sa = __builtin_amdgcn_wmma_f32_16x16x32_bf16(
              false, qf[c], false, kf, (short)0, sa, false, false);
        }
        int kg = kb + ti * 16 + ln;             // key index for this lane
#pragma unroll
        for (int r = 0; r < 8; ++r) {
          int qg = q0 + 8 * lh + r;
          sa[r] = (kg <= qg) ? sa[r] * scale : -1e30f;
        }
        s[ti] = sa;
      }

      // ---- online softmax over these 32 keys ----
#pragma unroll
      for (int r = 0; r < 8; ++r) {
        float rm = fmaxf(s[0][r], s[1][r]);
        rm = fmaxf(rm, __shfl_xor(rm, 1));
        rm = fmaxf(rm, __shfl_xor(rm, 2));
        rm = fmaxf(rm, __shfl_xor(rm, 4));
        rm = fmaxf(rm, __shfl_xor(rm, 8));
        float nM    = fmaxf(Mr[r], rm);
        float alpha = __expf(Mr[r] - nM);
        float p0 = __expf(s[0][r] - nM);
        float p1 = __expf(s[1][r] - nM);
        float rs = p0 + p1;
        rs += __shfl_xor(rs, 1);
        rs += __shfl_xor(rs, 2);
        rs += __shfl_xor(rs, 4);
        rs += __shfl_xor(rs, 8);
        Lr[r] = Lr[r] * alpha + rs;
        Mr[r] = nM;
#pragma unroll
        for (int dt = 0; dt < 8; ++dt) oacc[dt][r] *= alpha;
        // stage P (C-layout -> LDS) for the A-layout reload
        Ps[wave][8 * lh + r][ln]      = (__bf16)p0;
        Ps[wave][8 * lh + r][16 + ln] = (__bf16)p1;
      }
      asm volatile("s_wait_dscnt 0" ::: "memory");  // wave-private transpose

      // ---- P (16x32, A-layout) ----
      v16bf pf;
      {
        v8bf lo = *(const v8bf*)&Ps[wave][ln][lh * 8];
        v8bf hi = *(const v8bf*)&Ps[wave][ln][16 + lh * 8];
#pragma unroll
        for (int e = 0; e < 8; ++e) { pf[e] = lo[e]; pf[e + 8] = hi[e]; }
      }
      // ---- O += P * V ----
#pragma unroll
      for (int dt = 0; dt < 8; ++dt) {
        v8bf lo = *(const v8bf*)&Vt[dt * 16 + ln][lh * 8];
        v8bf hi = *(const v8bf*)&Vt[dt * 16 + ln][16 + lh * 8];
        v16bf vf;
#pragma unroll
        for (int e = 0; e < 8; ++e) { vf[e] = lo[e]; vf[e + 8] = hi[e]; }
        oacc[dt] = __builtin_amdgcn_wmma_f32_16x16x32_bf16(
            false, pf, false, vf, (short)0, oacc[dt], false, false);
      }
    }
  }

  // epilogue: normalize, write [B*T][E] bf16 (E index = h*HD + d)
  const int b = bh / H_, h = bh % H_;
#pragma unroll
  for (int dt = 0; dt < 8; ++dt) {
#pragma unroll
    for (int r = 0; r < 8; ++r) {
      int qrow = q0 + 8 * lh + r;
      size_t o = (size_t)(b * T_ + qrow) * E_ + h * HD_ + dt * 16 + ln;
      O[o] = (__bf16)(oacc[dt][r] / Lr[r]);
    }
  }
}

// ---------------------------------------------------------------------------
// launch
// ---------------------------------------------------------------------------
extern "C" void kernel_launch(void* const* d_in, const int* in_sizes, int n_in,
                              void* d_out, int out_size, void* d_ws, size_t ws_size,
                              hipStream_t stream) {
  const float* x     = (const float*)d_in[0];
  const float* freqs = (const float*)d_in[1];
  const float* w_qkv = (const float*)d_in[2];
  const float* w_o   = (const float*)d_in[3];
  float* out = (float*)d_out;

  // workspace carve-out (~302 MB total), 256B aligned
  char* ws = (char*)d_ws;
  size_t off = 0;
  auto carve = [&](size_t bytes) {
    void* p = ws + off;
    off = (off + bytes + 255) & ~(size_t)255;
    return p;
  };
  const size_t BTE = (size_t)B_ * T_ * E_;
  __bf16* xb    = (__bf16*)carve(BTE * 2);
  __bf16* wqkvb = (__bf16*)carve((size_t)3 * E_ * E_ * 2);
  __bf16* wob   = (__bf16*)carve((size_t)E_ * E_ * 2);
  __bf16* qkvb  = (__bf16*)carve(BTE * 3 * 2);
  __bf16* qbuf  = (__bf16*)carve(BTE * 2);
  __bf16* kbuf  = (__bf16*)carve(BTE * 2);
  __bf16* vbuf  = (__bf16*)carve(BTE * 2);
  __bf16* aout  = (__bf16*)carve(BTE * 2);

  // 1) fp32 -> bf16
  f32_to_bf16_kernel<<<4096, 256, 0, stream>>>(x, xb, (int)BTE);
  f32_to_bf16_kernel<<<4096, 256, 0, stream>>>(w_qkv, wqkvb, 3 * E_ * E_);
  f32_to_bf16_kernel<<<2048, 256, 0, stream>>>(w_o, wob, E_ * E_);

  // 2) qkv = x @ w_qkv^T   (M=8192, N=6144, K=2048)
  gemm_bf16_tn<__bf16><<<dim3(6144 / 128, 8192 / 128), 256, 0, stream>>>(
      xb, wqkvb, qkvb, B_ * T_, 3 * E_, E_);

  // 3) RoPE + split to [B][H][T][HD]
  rope_split_kernel<<<(B_ * H_ * T_ * 64) / 256, 256, 0, stream>>>(
      qkvb, freqs, qbuf, kbuf, vbuf);

  // 4) flash attention -> aout [B*T][E] bf16
  attn_kernel<<<dim3(T_ / 128, B_ * H_), 256, 0, stream>>>(qbuf, kbuf, vbuf, aout);

  // 5) out = aout @ w_o^T   (M=8192, N=2048, K=2048), fp32 output
  gemm_bf16_tn<float><<<dim3(2048 / 128, 8192 / 128), 256, 0, stream>>>(
      aout, wob, out, B_ * T_, E_, E_);
}
